// RevolutionNaiveOld_1039382085652
// MI455X (gfx1250) — matrix operand order repaired and checked by reading, not compiled, not run on hardware
//
#include <hip/hip_runtime.h>
#include <hip/hip_bf16.h>

#define HH 256
#define WW 256
#define BB 2
#define CC 64
#define GCH 16
#define KIN 420      // conv1 input features
#define KTILES 14    // 448 / 32
#define COUT1 16
#define PXW 64       // pixels per workgroup
#define NTHR 128     // 4 wave32

#define PATCH_RS 68            // LDS patch row stride (cols x0-1 .. x0+64, padded)
#define PATCH_CS (3 * 68)      // per-channel stride = 204
#define FRS 456                // feature-matrix row stride (bf16), 420 padded
#define W1_RS 456              // conv1 weight row stride (bf16)

// ---- single LDS pool with overlay: {gmax,rmax} (pre-GEMM) alias {sY1,sT,sSM} (post-GEMM)
#define OFF_PATCH 0            // float[64*204]            52224 B
#define OFF_F     52224        // ushort[64*456]           58368 B
#define OFF_W1    110592       // ushort[16*456]           14592 B
#define OFF_GMAX  125184       // float[4*3*68]             3264 B  (overlay A)
#define OFF_RMAX  128448       // float[64*68]             17408 B  (overlay A)
#define OFF_Y1    125184       // float[64*16]              4096 B  (overlay B)
#define OFF_T     129280       // float[64*16]              4096 B  (overlay B)
#define OFF_SM    133376       // float[64*36]              9216 B  (overlay B)
#define OFF_W2C   145856       // float[256]
#define OFF_W3    146880       // float[576]
#define OFF_B1    149184       // float[16]
#define OFF_B2    149248       // float[16]
#define OFF_B3    149312       // float[36]
#define OFF_SCALE 149456       // float[16]
#define OFF_SHIFT 149520       // float[16]
#define POOLSZ    149600

typedef __attribute__((ext_vector_type(16))) __bf16        v16bf;
typedef __attribute__((ext_vector_type(8)))  float         v8f;
typedef __attribute__((ext_vector_type(8)))  unsigned int  v8u;

__device__ __forceinline__ unsigned bf16bits(float f) {
  unsigned u = __float_as_uint(f);
  return (u + 0x7FFFu + ((u >> 16) & 1u)) >> 16;   // round-to-nearest-even
}
__device__ __forceinline__ float mishf(float x) {
  float sp = (x > 20.f) ? x : log1pf(__expf(x));
  return x * tanhf(sp);
}

__global__ __launch_bounds__(NTHR) void revolution_kernel(
    const float* __restrict__ x,
    const float* __restrict__ w1, const float* __restrict__ b1,
    const float* __restrict__ w2, const float* __restrict__ b2,
    const float* __restrict__ w3, const float* __restrict__ b3,
    const float* __restrict__ gma, const float* __restrict__ bta,
    const float* __restrict__ mu,  const float* __restrict__ var,
    float* __restrict__ out)
{
  __shared__ __align__(16) unsigned char sPool[POOLSZ];
  float*          sPatch = (float*)(sPool + OFF_PATCH);
  unsigned short* sF     = (unsigned short*)(sPool + OFF_F);
  unsigned short* sW1    = (unsigned short*)(sPool + OFF_W1);
  float*          sGmax  = (float*)(sPool + OFF_GMAX);
  float*          sRmax  = (float*)(sPool + OFF_RMAX);
  float*          sY1    = (float*)(sPool + OFF_Y1);
  float*          sT     = (float*)(sPool + OFF_T);
  float*          sSM    = (float*)(sPool + OFF_SM);
  float*          sW2c   = (float*)(sPool + OFF_W2C);
  float*          sW3    = (float*)(sPool + OFF_W3);
  float*          sB1    = (float*)(sPool + OFF_B1);
  float*          sB2    = (float*)(sPool + OFF_B2);
  float*          sB3    = (float*)(sPool + OFF_B3);
  float*          sScale = (float*)(sPool + OFF_SCALE);
  float*          sShift = (float*)(sPool + OFF_SHIFT);

  const int t  = threadIdx.x;
  const int wg = blockIdx.x;
  const int b  = wg >> 10;          // / (256 rows * 4 tiles)
  const int r  = wg & 1023;
  const int y  = r >> 2;
  const int x0 = (r & 3) << 6;

  // ---- Phase A: stage 3 input rows (y-1..y+1), cols x0-1..x0+64, all 64 ch ----
  for (int i = t; i < CC * PATCH_CS; i += NTHR) {
    int c   = i / PATCH_CS;
    int rem = i - c * PATCH_CS;
    int row = rem / PATCH_RS;
    int col = rem - row * PATCH_RS;
    int gy = y + row - 1;
    int gx = x0 + col - 1;
    float v = 0.f;
    if (col < 66 && gy >= 0 && gy < HH && gx >= 0 && gx < WW)
      v = x[(((size_t)b * CC + c) * HH + gy) * WW + gx];
    sPatch[i] = v;
  }
  // ---- Phase B: conv1 weights -> bf16 (K-padded) + small params -> LDS ----
  for (int i = t; i < COUT1 * W1_RS; i += NTHR) {
    int o = i / W1_RS, k = i - o * W1_RS;
    float v = (k < KIN) ? w1[o * KIN + k] : 0.f;
    sW1[i] = (unsigned short)bf16bits(v);
  }
  for (int i = t; i < 256; i += NTHR) sW2c[i] = w2[i * 9 + 4];   // center tap only
  for (int i = t; i < 576; i += NTHR) sW3[i] = w3[i];
  if (t < 16) {
    sB1[t] = b1[t];
    sB2[t] = b2[t];
    float sc = gma[t] * rsqrtf(var[t] + 1e-5f);
    sScale[t] = sc;
    sShift[t] = bta[t] - mu[t] * sc;                             // BN folded
  }
  if (t < 36) sB3[t] = b3[t];
  __syncthreads();

  // ---- Phase P1: shared max planes (factor out redundant reductions) ----
  // gmax[g][ki][col] = max over 16 group channels
  for (int i = t; i < 12 * PATCH_RS; i += NTHR) {
    int rowidx = i / PATCH_RS;                 // g*3 + ki
    int col    = i - rowidx * PATCH_RS;
    int g  = rowidx / 3, ki = rowidx - 3 * g;
    const float* p = sPatch + (g * GCH) * PATCH_CS + ki * PATCH_RS + col;
    float m = p[0];
#pragma unroll
    for (int c = 1; c < GCH; ++c) m = fmaxf(m, p[c * PATCH_CS]);
    sGmax[i] = m;
  }
  // rmax[c][col] = max over the 3 rows
  for (int i = t; i < CC * PATCH_RS; i += NTHR) {
    int c   = i / PATCH_RS;
    int col = i - c * PATCH_RS;
    const float* p = sPatch + c * PATCH_CS + col;
    sRmax[i] = fmaxf(fmaxf(p[0], p[PATCH_RS]), p[2 * PATCH_RS]);
  }
  __syncthreads();

  // ---- Phase F1: m_gc -> F cols [0,36): 1-load gather from gmax ----
  for (int i = t; i < 36 * PXW; i += NTHR) {
    int px = i & (PXW - 1);
    int f  = i >> 6;                         // g*9 + ki*3 + kj
    int g  = f / 9, kk = f - 9 * g;
    int ki = kk / 3, kj = kk - 3 * ki;
    float m = sGmax[(g * 3 + ki) * PATCH_RS + px + kj];
    sF[px * FRS + f] = (unsigned short)bf16bits(m);
  }
  // ---- Phase F2: m_kh -> F cols [36,228): 1-load gather from rmax ----
  for (int i = t; i < 192 * PXW; i += NTHR) {
    int px = i & (PXW - 1);
    int j  = i >> 6;                         // c*3 + kw
    int c  = j / 3, kw = j - 3 * c;
    float m = sRmax[c * PATCH_RS + px + kw];
    sF[px * FRS + 36 + j] = (unsigned short)bf16bits(m);
  }
  // ---- Phase F3: m_kw -> F cols [228,420): direct 3-col max (window slides w/ px) ----
  for (int i = t; i < 192 * PXW; i += NTHR) {
    int px = i & (PXW - 1);
    int j  = i >> 6;                         // c*3 + kh
    int c  = j / 3, kh = j - 3 * c;
    const float* p = sPatch + c * PATCH_CS + kh * PATCH_RS + px;
    float m = fmaxf(fmaxf(p[0], p[1]), p[2]);
    sF[px * FRS + 228 + j] = (unsigned short)bf16bits(m);
  }
  // ---- K padding: F cols [420,456) = 0 ----
  for (int i = t; i < 36 * PXW; i += NTHR) {
    int px = i & (PXW - 1);
    int f  = i >> 6;
    sF[px * FRS + 420 + f] = 0;
  }
  __syncthreads();     // after this, overlay region switches to sY1/sT/sSM

  const int lane = t & 31;
  const int wv   = t >> 5;
  const int s    = lane >> 4;       // half-wave select
  const int ln   = lane & 15;
  const int pxb  = wv * 16;         // this wave's 16-pixel M-tile

  // ---- Phase C: conv1 GEMM (M=16 px, N=16, K=448) via v_wmma_f32_16x16x32_bf16 ----
  // A frag: lane row = pixel; lanes 0-15 hold K {0-7,16-23}, lanes 16-31 K {8-15,24-31}
  // B frag: lane = output column; lanes 0-15 hold K {0-7,16-23}, lanes 16-31 K {8-15,24-31}
  v8f acc = {0.f, 0.f, 0.f, 0.f, 0.f, 0.f, 0.f, 0.f};
  const unsigned short* fA = &sF[(pxb + ln) * FRS];
  const unsigned short* fB = &sW1[ln * W1_RS];
#pragma unroll
  for (int kt = 0; kt < KTILES; ++kt) {
    const int k0 = kt * 32;
    uint4 a0 = *reinterpret_cast<const uint4*>(fA + k0 + s * 8);
    uint4 a1 = *reinterpret_cast<const uint4*>(fA + k0 + 16 + s * 8);
    uint4 b0 = *reinterpret_cast<const uint4*>(fB + k0 + s * 8);
    uint4 b1 = *reinterpret_cast<const uint4*>(fB + k0 + 16 + s * 8);
    v8u au, bu;
    au[0] = a0.x; au[1] = a0.y; au[2] = a0.z; au[3] = a0.w;
    au[4] = a1.x; au[5] = a1.y; au[6] = a1.z; au[7] = a1.w;
    bu[0] = b0.x; bu[1] = b0.y; bu[2] = b0.z; bu[3] = b0.w;
    bu[4] = b1.x; bu[5] = b1.y; bu[6] = b1.z; bu[7] = b1.w;
    v16bf Am = __builtin_bit_cast(v16bf, au);
    v16bf Bm = __builtin_bit_cast(v16bf, bu);
    acc = __builtin_amdgcn_wmma_f32_16x16x32_bf16(false, Am, false, Bm,
                                                  (short)0, acc, false, false);
  }

  // ---- Phase D: bias + Mish; scatter D tile (lane=N, vgpr=M) to sY1 ----
  {
    float bias = sB1[ln];
#pragma unroll
    for (int v = 0; v < 8; ++v) {
      int pxm = v + 8 * s;
      sY1[(pxb + pxm) * COUT1 + ln] = mishf(acc[v] + bias);
    }
  }
  __syncthreads();

  // ---- Phase E: conv2 center tap + folded BN + mish gate ----
  {
    int px = pxb + (lane >> 1);
    int o0 = (lane & 1) * 8;
    float yv[16];
#pragma unroll
    for (int i = 0; i < 16; ++i) yv[i] = sY1[px * COUT1 + i];
#pragma unroll
    for (int oo = 0; oo < 8; ++oo) {
      int o = o0 + oo;
      float a2 = sB2[o];
#pragma unroll
      for (int i = 0; i < 16; ++i) a2 += sW2c[o * 16 + i] * yv[i];
      float z = a2 * sScale[o] + sShift[o];
      sT[px * COUT1 + o] = yv[o] * mishf(z);
    }
  }
  __syncthreads();

  // ---- Phase F4: conv3 (16 -> 36) + per-group softmax over 9 taps ----
  {
    int px = pxb + ln;
    float tv[16];
#pragma unroll
    for (int i = 0; i < 16; ++i) tv[i] = sT[px * COUT1 + i];
#pragma unroll
    for (int gi = 0; gi < 2; ++gi) {
      int g = 2 * s + gi;
      float lg[9];
      float mx = -3.0e38f;
#pragma unroll
      for (int kk = 0; kk < 9; ++kk) {
        float a3 = sB3[g * 9 + kk];
#pragma unroll
        for (int i = 0; i < 16; ++i) a3 += sW3[(g * 9 + kk) * 16 + i] * tv[i];
        lg[kk] = a3;
        mx = fmaxf(mx, a3);
      }
      float sum = 0.f;
#pragma unroll
      for (int kk = 0; kk < 9; ++kk) { lg[kk] = __expf(lg[kk] - mx); sum += lg[kk]; }
      float inv = 1.f / sum;
#pragma unroll
      for (int kk = 0; kk < 9; ++kk) sSM[px * 36 + g * 9 + kk] = lg[kk] * inv;
    }
  }
  __syncthreads();

  // ---- Phase G: softmax-weighted 3x3 aggregation, coalesced channel-planar store ----
  {
    int px = t & (PXW - 1);
    int ch = t >> 6;                  // 0 or 1
    size_t obase = ((size_t)b * CC) * (HH * WW) + (size_t)y * WW + x0 + px;
#pragma unroll 1
    for (int it = 0; it < 32; ++it) {
      int c = 2 * it + ch;
      int g = c >> 4;
      const float* pp = sPatch + c * PATCH_CS + px;
      const float* sm = sSM + px * 36 + g * 9;
      float a = 0.f;
#pragma unroll
      for (int ki = 0; ki < 3; ++ki)
#pragma unroll
        for (int kj = 0; kj < 3; ++kj)
          a += pp[ki * PATCH_RS + kj] * sm[ki * 3 + kj];
      out[obase + (size_t)c * (HH * WW)] = a;
    }
  }
}

extern "C" void kernel_launch(void* const* d_in, const int* in_sizes, int n_in,
                              void* d_out, int out_size, void* d_ws, size_t ws_size,
                              hipStream_t stream) {
  const float* x   = (const float*)d_in[0];
  const float* w1  = (const float*)d_in[1];
  const float* b1  = (const float*)d_in[2];
  const float* w2  = (const float*)d_in[3];
  const float* b2  = (const float*)d_in[4];
  const float* w3  = (const float*)d_in[5];
  const float* b3  = (const float*)d_in[6];
  const float* gma = (const float*)d_in[7];
  const float* bta = (const float*)d_in[8];
  const float* mu  = (const float*)d_in[9];
  const float* var = (const float*)d_in[10];
  (void)in_sizes; (void)n_in; (void)out_size; (void)d_ws; (void)ws_size;

  dim3 grid(BB * HH * (WW / PXW));   // 2 * 256 * 4 = 2048 workgroups
  dim3 block(NTHR);                  // 4 wave32 per workgroup
  revolution_kernel<<<grid, block, 0, stream>>>(x, w1, b1, w2, b2, w3, b3,
                                                gma, bta, mu, var, (float*)d_out);
}
